// MDLSTM_CTC_7121055777334
// MI455X (gfx1250) — matrix operand order repaired and checked by reading, not compiled, not run on hardware
//
#include <hip/hip_runtime.h>

// MDLSTM + height-collapse + FC + log-softmax + CTC for MI455X (gfx1250).
//
//  k_init      : zero cross-row progress flags
//  k_pack_wh   : fp32 Wh1/Wh2 -> fp16 WMMA B-fragments (K=256,N=640)
//  k_pack_wfc  : fp32 W_fc    -> fp16 WMMA B-fragments (K=128,N=112 zero-padded)
//  k_mdlstm    : 32 persistent workgroups (1 per image row), wavefront-pipelined over
//                256 columns. Per cell: [32x256]x[256x640] GEMM via v_wmma_f32_16x16x32_f16.
//                Wave w owns hid-slice n in [16w,16w+16) of all 5 gates -> z and c stay in
//                VGPRs; h goes through 8KB LDS (A operand). 280KB of packed weights staged
//                in dynamic LDS (ds_load_b128), last K-group streamed from L2.
//  k_reduce    : sum h over height -> feat16 [8192][128] fp16
//  k_fc        : WMMA GEMM feat16 @ W_fc + b_fc -> logits [8192][112] f32
//  k_ctc       : per-batch block: log-softmax + 65-state CTC forward DP in LDS -> loss[32]

typedef __attribute__((ext_vector_type(16))) _Float16 v16h;
typedef __attribute__((ext_vector_type(8)))  _Float16 v8h;
typedef __attribute__((ext_vector_type(8)))  float    v8f;

#define NEGV (-1e30f)

constexpr int Bsz = 32;     // batch
constexpr int Tn  = 256;    // width (time)
constexpr int Fn  = 32;     // height (rows)
constexpr int HIDc = 128;
constexpr int ZN  = 5 * HIDc;   // 640
constexpr int VOC = 101;        // VOCAB+1 (blank = 100)
constexpr int NPAD = 112;       // 101 padded to 7*16
constexpr int LBL = 32;
constexpr int S_CTC = 2 * LBL + 1;  // 65
constexpr int KTL = 7;          // K-groups of Wh staged in LDS (7*40KB = 280KB)

__device__ __forceinline__ float sigm(float x) { return 1.f / (1.f + __expf(-x)); }

// ---- WMMA fragment loaders (16-bit A 16x32 layout per CDNA5 ISA 7.12.2) ----
// lane 0-15 : M=lane,    VGPR0-3 = K k0..k0+7 (k0=0),  VGPR4-7 = K k0+16..k0+23
// lane 16-31: M=lane-16, same with k0=8
__device__ __forceinline__ v16h load_a_frag(const _Float16* base, int ld, int mtile,
                                            int kbase, int lane) {
  int row = mtile * 16 + (lane & 15);
  int k0  = kbase + ((lane >> 4) << 3);
  const _Float16* p = base + row * ld + k0;
  v8h lo = *(const v8h*)(p);
  v8h hi = *(const v8h*)(p + 16);
  v16h out;
#pragma unroll
  for (int t = 0; t < 8; ++t) { out[t] = lo[t]; out[8 + t] = hi[t]; }
  return out;
}

// B fragments pre-packed: frag = 512 f16 (1 KB): [chunk(2)][lane(32)][8 f16]
__device__ __forceinline__ v16h load_b_frag(const _Float16* frag, int lane) {
  v8h lo = *(const v8h*)(frag + lane * 8);
  v8h hi = *(const v8h*)(frag + 256 + lane * 8);
  v16h out;
#pragma unroll
  for (int t = 0; t < 8; ++t) { out[t] = lo[t]; out[8 + t] = hi[t]; }
  return out;
}

// ---------------------------------------------------------------- init flags
__global__ void k_init(int* progress) {
  if (threadIdx.x < Fn) progress[threadIdx.x] = 0;
}

// ------------------------------------------------- pack Wh1/Wh2 B-fragments
// B tile 32x16: lane<16: n=lane, k_local = chunk*16+p; lane>=16: n=lane-16, k_local+8.
__global__ void k_pack_wh(const float* Wh1, const float* Wh2, _Float16* WhPack) {
  int idx = blockIdx.x * 256 + threadIdx.x;          // 8*40*512 = 163840 elems
  if (idx >= 8 * 40 * 512) return;
  int frag = idx >> 9;
  int rem  = idx & 511;
  int c = rem >> 8, rem2 = rem & 255;
  int L = rem2 >> 3, p = rem2 & 7;
  int kt = frag / 40, nt = frag % 40;
  int k_local = ((L >> 4) << 3) + c * 16 + p;
  int K = kt * 32 + k_local;
  int N = nt * 16 + (L & 15);
  float v = (K < HIDc) ? Wh1[K * ZN + N] : Wh2[(K - HIDc) * ZN + N];
  WhPack[idx] = (_Float16)v;
}

__global__ void k_pack_wfc(const float* Wfc, _Float16* WfcPack) {
  int idx = blockIdx.x * 256 + threadIdx.x;          // 4*7*512 = 14336 elems
  if (idx >= 4 * 7 * 512) return;
  int frag = idx >> 9;
  int rem  = idx & 511;
  int c = rem >> 8, rem2 = rem & 255;
  int L = rem2 >> 3, p = rem2 & 7;
  int kt = frag / 7, nt = frag % 7;
  int k_local = ((L >> 4) << 3) + c * 16 + p;
  int K = kt * 32 + k_local;
  int N = nt * 16 + (L & 15);
  float v = (N < VOC) ? Wfc[K * VOC + N] : 0.f;
  WfcPack[idx] = (_Float16)v;
}

// -------------------------------------------------------- MDLSTM row kernel
// grid = 32 blocks (one per row i), block = 256 threads = 8 waves.
__global__ __launch_bounds__(256, 1)
void k_mdlstm(const float* __restrict__ X, const float* __restrict__ Wx,
              const float* __restrict__ bias, const _Float16* __restrict__ WhPack,
              _Float16* __restrict__ h_glob, _Float16* __restrict__ c_glob,
              int* progress) {
  __shared__ __align__(16) _Float16 hl[Bsz * HIDc];   // own-row h (A src, K 0..127)
  __shared__ float x_s[Bsz];
  extern __shared__ __align__(16) char smem_dyn[];    // KTL*40*512 f16 of WhPack
  _Float16* whl = (_Float16*)smem_dyn;

  const int row  = blockIdx.x;
  const int tid  = threadIdx.x;
  const int lane = tid & 31;
  const int w    = tid >> 5;           // wave id: owns hid-slice [16w, 16w+16)
  const int nl   = w * 16 + (lane & 15);  // hid index owned by this lane

  // per-lane gate constants: Wx/bias at N = g*128 + nl
  float wx_g[5], b_g[5];
#pragma unroll
  for (int g = 0; g < 5; ++g) { wx_g[g] = Wx[g * HIDc + nl]; b_g[g] = bias[g * HIDc + nl]; }

  // stage K-groups 0..KTL-1 of packed weights into LDS (16B vector copies)
  {
    const v8h* src = (const v8h*)WhPack;
    v8h* dst = (v8h*)whl;
    for (int e = tid; e < KTL * 40 * 64; e += 256) dst[e] = src[e];
  }
  for (int e = tid; e < Bsz * HIDc; e += 256) hl[e] = (_Float16)0.f;

  float c_reg[2][8];                   // cell state, resident in VGPRs
#pragma unroll
  for (int m = 0; m < 2; ++m)
#pragma unroll
    for (int r = 0; r < 8; ++r) c_reg[m][r] = 0.f;
  __syncthreads();

  for (int j = 0; j < Tn; ++j) {
    // wait for row above to finish column j (agent-scope acquire)
    if (row > 0) {
      if (tid == 0) {
        while (__hip_atomic_load(&progress[row - 1], __ATOMIC_ACQUIRE,
                                 __HIP_MEMORY_SCOPE_AGENT) <= j) {
          __builtin_amdgcn_s_sleep(1);
        }
      }
      __syncthreads();
    }
    if (tid < Bsz) x_s[tid] = X[tid * (Tn * Fn) + j * Fn + row];  // X[b][j][row]

    // opaque zero offsets: defeat loop-invariant hoisting (-> scratch spills) of
    // the weight-fragment loads; they must stay in-loop as ds/global loads.
    int od = 0, og = 0;
    asm volatile("" : "+v"(od), "+s"(og));
    const _Float16* whl_j = whl + od;
    const _Float16* whg_j = WhPack + og;

    const _Float16* hu = h_glob + (size_t)((row - 1) * Tn + j) * (Bsz * HIDc);

    // ---- WMMA phase: z = [hl ; hu] @ [Wh1;Wh2], acc[g][m] in registers ----
    v8f acc[5][2] = {};
    const int ktmax = (row > 0) ? 8 : 4;   // top row: h_up == 0
    for (int kt = 0; kt < ktmax; ++kt) {
      v16h bfr[5];
      if (kt < KTL) {
#pragma unroll
        for (int g = 0; g < 5; ++g)
          bfr[g] = load_b_frag(whl_j + (size_t)(kt * 40 + g * 8 + w) * 512, lane);
      } else {
#pragma unroll
        for (int g = 0; g < 5; ++g)
          bfr[g] = load_b_frag(whg_j + (size_t)(kt * 40 + g * 8 + w) * 512, lane);
      }
      v16h a0, a1;
      if (kt < 4) {
        a0 = load_a_frag(hl, HIDc, 0, kt * 32, lane);
        a1 = load_a_frag(hl, HIDc, 1, kt * 32, lane);
      } else {
        a0 = load_a_frag(hu, HIDc, 0, (kt - 4) * 32, lane);
        a1 = load_a_frag(hu, HIDc, 1, (kt - 4) * 32, lane);
      }
#pragma unroll
      for (int g = 0; g < 5; ++g)
        acc[g][0] = __builtin_amdgcn_wmma_f32_16x16x32_f16(
            false, a0, false, bfr[g], (short)0, acc[g][0], false, false);
#pragma unroll
      for (int g = 0; g < 5; ++g)
        acc[g][1] = __builtin_amdgcn_wmma_f32_16x16x32_f16(
            false, a1, false, bfr[g], (short)0, acc[g][1], false, false);
    }
    __syncthreads();   // all A-reads of hl complete before gate phase rewrites it

    // ---- gate phase: z and c in registers, h -> LDS + global ----
    const _Float16* cu = c_glob + (size_t)((row - 1) * Tn + j) * (Bsz * HIDc);
    _Float16* hout = h_glob + (size_t)(row * Tn + j) * (Bsz * HIDc);
    _Float16* cout = c_glob + (size_t)(row * Tn + j) * (Bsz * HIDc);
#pragma unroll
    for (int m = 0; m < 2; ++m) {
#pragma unroll
      for (int r = 0; r < 8; ++r) {
        int brow = m * 16 + r + ((lane >> 4) << 3);   // batch index (C-layout row)
        float xs = x_s[brow];
        float zi = acc[0][m][r] + xs * wx_g[0] + b_g[0];
        float zg = acc[1][m][r] + xs * wx_g[1] + b_g[1];
        float z1 = acc[2][m][r] + xs * wx_g[2] + b_g[2];
        float z2 = acc[3][m][r] + xs * wx_g[3] + b_g[3];
        float zo = acc[4][m][r] + xs * wx_g[4] + b_g[4];
        float cuv = (row > 0) ? (float)cu[brow * HIDc + nl] : 0.f;
        float c = sigm(z1) * c_reg[m][r] + sigm(z2) * cuv + sigm(zi) * tanhf(zg);
        float h = sigm(zo) * tanhf(c);
        c_reg[m][r] = c;
        hl[brow * HIDc + nl]   = (_Float16)h;
        hout[brow * HIDc + nl] = (_Float16)h;
        cout[brow * HIDc + nl] = (_Float16)c;
      }
    }
    __threadfence();          // h/c stores agent-visible before flag
    __syncthreads();
    if (tid == 0) {
      __hip_atomic_store(&progress[row], j + 1, __ATOMIC_RELEASE,
                         __HIP_MEMORY_SCOPE_AGENT);
    }
  }
}

// -------------------------------------------- height-collapse: feat16[m][n]
__global__ void k_reduce(const _Float16* __restrict__ h_glob, _Float16* __restrict__ feat16) {
  int idx = blockIdx.x * 256 + threadIdx.x;      // 8192*128 = 1,048,576
  if (idx >= Bsz * Tn * HIDc) return;
  int m = idx >> 7, n = idx & 127;
  int bb = m >> 8, t = m & 255;
  float s = 0.f;
#pragma unroll 4
  for (int i = 0; i < Fn; ++i)
    s += (float)h_glob[((size_t)(i * Tn + t) * Bsz + bb) * HIDc + n];
  feat16[idx] = (_Float16)s;
}

// ------------------------------------------------- FC GEMM: 8192x112 = feat @ Wfc
__global__ __launch_bounds__(256)
void k_fc(const _Float16* __restrict__ feat16, const _Float16* __restrict__ WfcPack,
          const float* __restrict__ b_fc, float* __restrict__ logits) {
  int lane  = threadIdx.x & 31;
  int wave  = threadIdx.x >> 5;
  int mtile = blockIdx.x * 8 + wave;   // 512 total M tiles
  v16h a[4];
#pragma unroll
  for (int kt = 0; kt < 4; ++kt) a[kt] = load_a_frag(feat16, HIDc, mtile, kt * 32, lane);
#pragma unroll
  for (int nt = 0; nt < 7; ++nt) {
    v8f acc = {};
#pragma unroll
    for (int kt = 0; kt < 4; ++kt) {
      v16h bb = load_b_frag(WfcPack + (size_t)(kt * 7 + nt) * 512, lane);
      acc = __builtin_amdgcn_wmma_f32_16x16x32_f16(false, a[kt], false, bb, (short)0,
                                                   acc, false, false);
    }
    int n = nt * 16 + (lane & 15);
    float bias = (n < VOC) ? b_fc[n] : 0.f;
    int rbase = mtile * 16 + ((lane >> 4) << 3);
#pragma unroll
    for (int r = 0; r < 8; ++r) logits[(size_t)(rbase + r) * NPAD + n] = acc[r] + bias;
  }
}

// --------------------------------- CTC: log-softmax + forward DP, 1 block / batch
__global__ __launch_bounds__(256)
void k_ctc(const float* __restrict__ logits, const int* __restrict__ y,
           float* __restrict__ loss) {
  __shared__ float lp[Tn][S_CTC + 1];     // lp_ext[t][s], padded stride
  __shared__ int   ext[S_CTC];
  __shared__ unsigned char skipf[S_CTC];
  __shared__ float alpha[2][S_CTC + 1];

  const int b = blockIdx.x, tid = threadIdx.x;
  if (tid < S_CTC) ext[tid] = (tid & 1) ? y[b * LBL + (tid >> 1)] : (VOC - 1);
  __syncthreads();
  if (tid < S_CTC)
    skipf[tid] = (tid >= 2) && (ext[tid] != (VOC - 1)) && (ext[tid] != ext[tid - 2]);

  {
    int t = tid;  // 256 threads, 256 time steps
    const float* lrow = logits + (size_t)(b * Tn + t) * NPAD;
    float mx = NEGV;
    for (int v = 0; v < VOC; ++v) mx = fmaxf(mx, lrow[v]);
    float s = 0.f;
    for (int v = 0; v < VOC; ++v) s += __expf(lrow[v] - mx);
    float lse = mx + __logf(s);
    for (int ss = 0; ss < S_CTC; ++ss) lp[t][ss] = lrow[ext[ss]] - lse;
  }
  __syncthreads();

  if (tid < S_CTC) alpha[0][tid] = NEGV;
  __syncthreads();
  if (tid == 0) alpha[0][0] = lp[0][0];
  if (tid == 1) alpha[0][1] = lp[0][1];
  __syncthreads();

  int cur = 0;
  for (int t = 1; t < Tn; ++t) {
    if (tid < S_CTC) {
      float a0 = alpha[cur][tid];
      float a1 = (tid >= 1) ? alpha[cur][tid - 1] : NEGV;
      float a2 = (tid >= 2 && skipf[tid]) ? alpha[cur][tid - 2] : NEGV;
      float m = fmaxf(a0, fmaxf(a1, a2));
      float v;
      if (m <= -1e29f) v = NEGV;
      else v = m + __logf(__expf(a0 - m) + __expf(a1 - m) + __expf(a2 - m));
      alpha[cur ^ 1][tid] = v + lp[t][tid];
    }
    __syncthreads();
    cur ^= 1;
  }
  if (tid == 0) {
    float aN = alpha[cur][S_CTC - 1], aM = alpha[cur][S_CTC - 2];
    float m = fmaxf(aN, aM);
    loss[b] = -(m + __logf(__expf(aN - m) + __expf(aM - m)));
  }
}

// ------------------------------------------------------------------- launch
extern "C" void kernel_launch(void* const* d_in, const int* in_sizes, int n_in,
                              void* d_out, int out_size, void* d_ws, size_t ws_size,
                              hipStream_t stream) {
  const float* X    = (const float*)d_in[0];   // [32,256,32]
  const int*   y    = (const int*)  d_in[1];   // [32,32]
  const float* Wx   = (const float*)d_in[2];   // [1,640]
  const float* Wh1  = (const float*)d_in[3];   // [128,640]
  const float* Wh2  = (const float*)d_in[4];   // [128,640]
  const float* bias = (const float*)d_in[5];   // [640]
  const float* Wfc  = (const float*)d_in[6];   // [128,101]
  const float* bfc  = (const float*)d_in[7];   // [101]
  float* out = (float*)d_out;                  // [32]

  char* ws = (char*)d_ws;
  size_t off = 0;
  auto take = [&](size_t bytes) -> char* {
    char* p = ws + off;
    off = (off + bytes + 255) & ~(size_t)255;
    return p;
  };
  int*       progress = (int*)      take(Fn * sizeof(int));
  _Float16*  WhPack   = (_Float16*) take((size_t)8 * 40 * 512 * 2);          // 320 KB
  _Float16*  WfcPack  = (_Float16*) take((size_t)4 * 7 * 512 * 2);           // 28 KB
  _Float16*  h_glob   = (_Float16*) take((size_t)Fn * Tn * Bsz * HIDc * 2);  // 64 MB
  _Float16*  c_glob   = (_Float16*) take((size_t)Fn * Tn * Bsz * HIDc * 2);  // 64 MB
  _Float16*  feat16   = (_Float16*) take((size_t)Bsz * Tn * HIDc * 2);       // 2 MB
  float*     logits   = (float*)    take((size_t)Bsz * Tn * NPAD * 4);       // 3.5 MB

  const size_t dynLds = (size_t)KTL * 40 * 512 * sizeof(_Float16);           // 280 KB

  k_init<<<1, 32, 0, stream>>>(progress);
  k_pack_wh<<<(8 * 40 * 512 + 255) / 256, 256, 0, stream>>>(Wh1, Wh2, WhPack);
  k_pack_wfc<<<(4 * 7 * 512 + 255) / 256, 256, 0, stream>>>(Wfc, WfcPack);
  k_mdlstm<<<Fn, 256, dynLds, stream>>>(X, Wx, bias, WhPack, h_glob, c_glob, progress);
  k_reduce<<<(Bsz * Tn * HIDc + 255) / 256, 256, 0, stream>>>(h_glob, feat16);
  k_fc<<<64, 256, 0, stream>>>(feat16, WfcPack, bfc, logits);
  k_ctc<<<Bsz, 256, 0, stream>>>(logits, y, out);
}
